// GRU_90907277787184
// MI455X (gfx1250) — matrix operand order repaired and checked by reading, not compiled, not run on hardware
//
#include <hip/hip_runtime.h>

// ---------------------------------------------------------------------------
// GRU on MI455X (gfx1250, wave32, WMMA, TDM).
// Phase 1: wx = src @ w_ih^T + b_ih            (bf16 WMMA GEMM, whole GPU)
// Phase 2: persistent 16-WG scan, w_hh slices LDS-resident, bf16 WMMA/step,
//          h-state and wx staged into LDS via TENSOR_LOAD_TO_LDS (TDM)
// Phase 3: out = hs @ reg_w^T + reg_b          (bf16 WMMA GEMM, whole GPU)
// ---------------------------------------------------------------------------

typedef __attribute__((ext_vector_type(16))) __bf16 v16bf;
typedef __attribute__((ext_vector_type(8)))  __bf16 v8bf;
typedef __attribute__((ext_vector_type(8)))  float  v8f;
typedef __attribute__((ext_vector_type(4)))  unsigned int v4u;
typedef __attribute__((ext_vector_type(8)))  int    v8i_t;
typedef __attribute__((ext_vector_type(4)))  int    v4i_t;

#define DEVINL __device__ __forceinline__

// Problem dims
#define NB   32
#define TT   1024
#define CC   256
#define HH   512
#define OO   256
#define G3   1536
#define NWG  16      // workgroups in the scan
#define SCTH 384     // 12 waves per scan workgroup

#if __has_builtin(__builtin_amdgcn_tensor_load_to_lds)
#define HAVE_TDM 1
#endif

DEVINL __bf16 f2bf(float f) {
  unsigned u = __builtin_bit_cast(unsigned, f);
  u += 0x7FFFu + ((u >> 16) & 1u);          // round to nearest even
  unsigned short h = (unsigned short)(u >> 16);
  return __builtin_bit_cast(__bf16, h);
}

// A tile 16x32 bf16, row-major with leading dim ld (ISA 7.12.2 16-bit A layout)
DEVINL v16bf load_tileA(const __bf16* p, int ld) {
  const int l  = threadIdx.x & 31;
  const int m  = l & 15;
  const int kh = (l >> 4) << 3;             // 0 or 8
  const __bf16* r = p + m * ld;
  v8bf lo = *(const v8bf*)(r + kh);         // K = kh .. kh+7
  v8bf hi = *(const v8bf*)(r + 16 + kh);    // K = 16+kh .. 16+kh+7
  v16bf a;
#pragma unroll
  for (int i = 0; i < 8; ++i) { a[i] = lo[i]; a[8 + i] = hi[i]; }
  return a;
}

// B tile 32x16 (KxN): source is W row-major [Nrows][K] (so B = W^T), tile at
// W + n0*ld + k0.  Lane n = l&15 takes K = kh..kh+7 and 16+kh..16+kh+7 (as A).
DEVINL v16bf load_tileB(const __bf16* p, int ld) {
  const int l  = threadIdx.x & 31;
  const int n  = l & 15;
  const int kh = (l >> 4) << 3;             // 0 or 8
  const __bf16* r = p + n * ld + kh;
  v8bf lo = *(const v8bf*)(r);
  v8bf hi = *(const v8bf*)(r + 16);
  v16bf b;
#pragma unroll
  for (int i = 0; i < 8; ++i) { b[i] = lo[i]; b[8 + i] = hi[i]; }
  return b;
}

DEVINL v8f wmma_bf16(v16bf a, v16bf b, v8f c) {
  return __builtin_amdgcn_wmma_f32_16x16x32_bf16(
      /*neg_a=*/false, a, /*neg_b=*/false, b,
      /*c_mod=*/(short)0, c, /*reuse_a=*/false, /*reuse_b=*/false);
}

DEVINL float sigmoidf_fast(float x) { return 1.f / (1.f + __expf(-x)); }

// ---------------------------------------------------------------------------
// Elementwise f32 -> bf16 convert
__global__ void cvt_kernel(const float* __restrict__ in, __bf16* __restrict__ out, int n) {
  for (int i = blockIdx.x * blockDim.x + threadIdx.x; i < n; i += gridDim.x * blockDim.x)
    out[i] = f2bf(in[i]);
}

// Zero barrier counter + both h ping-pong buffers (fresh every launch/replay)
__global__ void init_kernel(unsigned* __restrict__ cnt, __bf16* __restrict__ hb) {
  int i = blockIdx.x * blockDim.x + threadIdx.x;
  if (i == 0) *cnt = 0u;
  if (i < 2 * NB * HH) hb[i] = __builtin_bit_cast(__bf16, (unsigned short)0);
}

// ---------------------------------------------------------------------------
// Generic C[M,N] = A[M,K](bf16,rm) * W[N,K](bf16,rm)^T + bias, f32 out.
// 256 threads = 8 waves, block tile 32x64 (2x4 waves of 16x16).
__global__ __launch_bounds__(256)
void gemm_bf16_kernel(const __bf16* __restrict__ A, const __bf16* __restrict__ W,
                      const float* __restrict__ bias, float* __restrict__ C,
                      int M, int N, int K) {
  const int w  = threadIdx.x >> 5;
  const int mi = w >> 2, nj = w & 3;
  const int m0 = blockIdx.y * 32 + mi * 16;
  const int n0 = blockIdx.x * 64 + nj * 16;
  const __bf16* Ap = A + (long)m0 * K;
  const __bf16* Wp = W + (long)n0 * K;
  v8f acc = {};
#pragma unroll 4
  for (int kb = 0; kb < K; kb += 32) {
    v16bf a = load_tileA(Ap + kb, K);
    v16bf b = load_tileB(Wp + kb, K);
    acc = wmma_bf16(a, b, acc);
  }
  const int l  = threadIdx.x & 31;
  const int n  = l & 15;
  const int mb = (l >> 4) << 3;
  const float bz = bias ? bias[n0 + n] : 0.f;
#pragma unroll
  for (int v = 0; v < 8; ++v)
    C[(long)(m0 + mb + v) * N + (n0 + n)] = acc[v] + bz;
}

// ---------------------------------------------------------------------------
// Persistent GRU scan.  16 WGs; WG wg owns h columns [32*wg, 32*wg+32) and the
// matching 96 rows of w_hh (r,z,n gates) resident in LDS.  h state ping-pongs
// through a 2x[32][512] bf16 global buffer; one agent-scope barrier per step.
// Per step, wave 0 issues two TENSOR_LOAD_TO_LDS DMAs: the 32KB h tile and
// this WG's wx gather (3D tile: 32 cols x 3 gates x 32 batch rows).
__global__ __launch_bounds__(SCTH)
void gru_scan_kernel(const __bf16* __restrict__ Whh,   // [1536][512] bf16
                     const float*  __restrict__ bhh,   // [1536]
                     const float*  __restrict__ wx,    // [32*1024][1536] f32
                     __bf16* __restrict__ hb,          // 2 x [32][512] bf16
                     __bf16* __restrict__ hs,          // [32*1024][512] bf16
                     unsigned* __restrict__ cnt) {
  extern __shared__ char smem[];
  __bf16* Wsl = (__bf16*)smem;                                    // 96*512 bf16 (96KB)
  __bf16* Hst = (__bf16*)(smem + 96 * 512 * 2);                   // 32*512 bf16 (32KB)
  float*  S   = (float*)(smem + 96 * 512 * 2 + 32 * 512 * 2);     // 32*96  f32 (12KB)
  float*  Hp  = (float*)(smem + 96 * 512 * 2 + 32 * 512 * 2 + 32 * 96 * 4);   // 32*32
  float*  WXs = (float*)(smem + 96 * 512 * 2 + 32 * 512 * 2 + 32 * 96 * 4 + 32 * 32 * 4); // 32*96

  const int tid = threadIdx.x;
  const int wg  = blockIdx.x;            // 0..15
  const int w   = tid >> 5;
  const int mi  = w / 6, nj = w % 6;     // 2 x 6 tiles covering [32][96]
  const int l   = tid & 31;
  const int n   = l & 15;
  const int mb  = (l >> 4) << 3;

  // Load this WG's 96x512 w_hh slice into LDS (rows g*512 + wg*32 + j).
  for (int i = tid; i < 96 * 64; i += SCTH) {
    int s  = i >> 6;                     // slice row 0..95
    int k8 = i & 63;                     // 8-elem chunk within row
    int G  = (s >> 5) * HH + wg * 32 + (s & 31);
    ((v8bf*)Wsl)[i] = ((const v8bf*)Whh)[(long)G * 64 + k8];
  }
  for (int i = tid; i < 32 * 32; i += SCTH) Hp[i] = 0.f;
  __syncthreads();

#ifdef HAVE_TDM
  // Tensor DMA descriptors (ISA 8.3/8.4).  Group1/2/3 are step-invariant.
  // h tile: 2D, data_size=2B, dim0=512, dim1=32, stride0=512.
  const v8i_t g1_h = { (int)0x00010000u,          // data_size=1 (2B)
                       (int)(512u << 16),         // tensor_dim0 = 512
                       (int)(32u << 16),          // tensor_dim1 = 32
                       (int)(512u << 16),         // tile_dim0 = 512
                       (int)32u,                  // tile_dim1 = 32
                       (int)512u,                 // tensor_dim0_stride = 512
                       0, 0 };
  // wx tile: 3D, data_size=4B, x=32 cols, y=3 gates (stride 512),
  // z=32 batch rows (stride T*3H = 1572864).
  const v8i_t g1_w = { (int)0x00020000u,          // data_size=2 (4B)
                       (int)(32u << 16),          // tensor_dim0 = 32
                       (int)(3u << 16),           // tensor_dim1 = 3
                       (int)(32u << 16),          // tile_dim0 = 32
                       (int)(3u | (32u << 16)),   // tile_dim1=3, tile_dim2=32
                       (int)512u,                 // dim0_stride = 512
                       (int)(((unsigned)(TT * G3) & 0xFFFFu) << 16), // dim1_stride lo16
                       (int)((unsigned)(TT * G3) >> 16) };           // dim1_stride hi
  const v4i_t g2_h = { 0, 0, 0, 0 };
  const v4i_t g2_w = { 32, 0, 0, 0 };             // tensor_dim2 = 32
  const v4i_t g3_z = { 0, 0, 0, 0 };
  const v8i_t g4_z = { 0, 0, 0, 0, 0, 0, 0, 0 };  // extra arg (clang-23 form)
  const unsigned lds_h  = (unsigned)(uintptr_t)Hst;   // low 32 bits = LDS addr
  const unsigned lds_wx = (unsigned)(uintptr_t)WXs;
#endif

  unsigned gen = 0;
  for (int t = 0; t < TT; ++t) {
    const __bf16* hbr = hb + (size_t)(t & 1) * (NB * HH);
    __bf16*       hbw = hb + (size_t)((t + 1) & 1) * (NB * HH);

#ifdef HAVE_TDM
    if (w == 0) {                       // one DMA issue per WG (EXEC ignored)
      unsigned long long gah = (unsigned long long)(uintptr_t)hbr;
      unsigned long long gaw = (unsigned long long)(uintptr_t)(wx + (long)t * G3 + wg * 32);
      v4u g0h = { 1u, lds_h,  (unsigned)gah, (unsigned)(gah >> 32) | 0x80000000u };
      v4u g0w = { 1u, lds_wx, (unsigned)gaw, (unsigned)(gaw >> 32) | 0x80000000u };
      __builtin_amdgcn_tensor_load_to_lds(g0h, g1_h, g2_h, g3_z, g4_z, 0);
      __builtin_amdgcn_tensor_load_to_lds(g0w, g1_w, g2_w, g3_z, g4_z, 0);
      __builtin_amdgcn_s_wait_tensorcnt(0);
    }
#else
    // Fallback staging: h (32x512 bf16) and wx chunk ([m][gate][j] f32).
    for (int i = tid; i < NB * HH / 8; i += SCTH)
      ((v8bf*)Hst)[i] = ((const v8bf*)hbr)[i];
    for (int i = tid; i < 32 * 96; i += SCTH) {
      int m = i / 96, gs = i % 96;
      WXs[i] = wx[(long)(m * TT + t) * G3 + (gs >> 5) * HH + wg * 32 + (gs & 31)];
    }
#endif
    __syncthreads();

    // wh = h @ Whh_slice^T : each wave one 16x16 f32 tile, K = 512.
    v8f acc = {};
#pragma unroll
    for (int kb = 0; kb < 16; ++kb) {
      v16bf a = load_tileA(Hst + mi * 16 * HH + kb * 32, HH);
      v16bf b = load_tileB(Wsl + nj * 16 * HH + kb * 32, HH);
      acc = wmma_bf16(a, b, acc);
    }
    {
      int s = nj * 16 + n;                               // slice col 0..95
      int G = (s >> 5) * HH + wg * 32 + (s & 31);        // global gate col
      float bz = bhh[G];
#pragma unroll
      for (int v = 0; v < 8; ++v)
        S[(mi * 16 + mb + v) * 96 + s] = acc[v] + bz;
    }
    __syncthreads();

    // Gates + state update for our 32 h-columns (32x32 elems).
    for (int e = tid; e < NB * 32; e += SCTH) {
      int m = e >> 5, j = e & 31;
      int col = wg * 32 + j;
      float wxr = WXs[(m * 3 + 0) * 32 + j];
      float wxz = WXs[(m * 3 + 1) * 32 + j];
      float wxn = WXs[(m * 3 + 2) * 32 + j];
      float Sr = S[m * 96 + j];
      float Sz = S[m * 96 + 32 + j];
      float Sn = S[m * 96 + 64 + j];
      float r  = sigmoidf_fast(wxr + Sr);
      float z  = sigmoidf_fast(wxz + Sz);
      float nn = tanhf(wxn + r * Sn);
      float hnew = (1.f - z) * nn + z * Hp[e];
      Hp[e] = hnew;
      __bf16 hv = f2bf(hnew);
      hbw[m * HH + col] = hv;
      hs[(long)(m * TT + t) * HH + col] = hv;
      if (t + 1 < TT)   // warm L2 for next step's TDM gather
        __builtin_prefetch(&wx[(long)(m * TT + t + 1) * G3 + col], 0, 0);
    }

    // Inter-WG barrier: release our h writes, acquire everyone else's.
    ++gen;
    __syncthreads();
    if (tid == 0) {
      __hip_atomic_fetch_add(cnt, 1u, __ATOMIC_RELEASE, __HIP_MEMORY_SCOPE_AGENT);
      const unsigned target = (unsigned)NWG * gen;
      while (__hip_atomic_load(cnt, __ATOMIC_ACQUIRE, __HIP_MEMORY_SCOPE_AGENT) < target)
        __builtin_amdgcn_s_sleep(1);
    }
    __syncthreads();
  }
}

// ---------------------------------------------------------------------------
extern "C" void kernel_launch(void* const* d_in, const int* in_sizes, int n_in,
                              void* d_out, int out_size, void* d_ws, size_t ws_size,
                              hipStream_t stream) {
  (void)in_sizes; (void)n_in; (void)out_size; (void)ws_size;
  const float* src  = (const float*)d_in[0];   // [32,1024,256]
  const float* w_ih = (const float*)d_in[1];   // [1536,256]
  const float* w_hh = (const float*)d_in[2];   // [1536,512]
  const float* b_ih = (const float*)d_in[3];   // [1536]
  const float* b_hh = (const float*)d_in[4];   // [1536]
  const float* regw = (const float*)d_in[5];   // [256,512]
  const float* regb = (const float*)d_in[6];   // [256]
  float* out = (float*)d_out;                  // [32,1024,256]

  char* ws = (char*)d_ws;
  size_t off = 0;
  auto alloc = [&](size_t bytes) -> char* {
    char* p = ws + off;
    off += (bytes + 255) & ~(size_t)255;
    return p;
  };
  float*    wx    = (float*)   alloc((size_t)NB * TT * G3 * 4);  // 192MB
  __bf16*   hsb   = (__bf16*)  alloc((size_t)NB * TT * HH * 2);  // 32MB
  __bf16*   srcb  = (__bf16*)  alloc((size_t)NB * TT * CC * 2);  // 16MB
  __bf16*   wihb  = (__bf16*)  alloc((size_t)G3 * CC * 2);
  __bf16*   whhb  = (__bf16*)  alloc((size_t)G3 * HH * 2);
  __bf16*   regwb = (__bf16*)  alloc((size_t)OO * HH * 2);
  __bf16*   hbuf  = (__bf16*)  alloc((size_t)2 * NB * HH * 2);
  unsigned* cnt   = (unsigned*)alloc(256);

  // bf16 conversions
  cvt_kernel<<<1024, 256, 0, stream>>>(src,  srcb,  NB * TT * CC);
  cvt_kernel<<<256,  256, 0, stream>>>(w_ih, wihb,  G3 * CC);
  cvt_kernel<<<512,  256, 0, stream>>>(w_hh, whhb,  G3 * HH);
  cvt_kernel<<<128,  256, 0, stream>>>(regw, regwb, OO * HH);
  init_kernel<<<(2 * NB * HH + 255) / 256, 256, 0, stream>>>(cnt, hbuf);

  // Phase 1: wx[32768,1536] = srcb[32768,256] @ wihb^T + b_ih
  gemm_bf16_kernel<<<dim3(G3 / 64, (NB * TT) / 32), 256, 0, stream>>>(
      srcb, wihb, b_ih, wx, NB * TT, G3, CC);

  // Phase 2: persistent scan.  Dynamic LDS: 96KB Whh + 32KB h stage + S + Hp
  // + 12KB wx stage = 159744B (CDNA5 WGP has 320KB).
  const int SMEM = 96 * 512 * 2 + 32 * 512 * 2 + 32 * 96 * 4 + 32 * 32 * 4 + 32 * 96 * 4;
  (void)hipFuncSetAttribute((const void*)gru_scan_kernel,
                            hipFuncAttributeMaxDynamicSharedMemorySize, SMEM);
  gru_scan_kernel<<<NWG, SCTH, SMEM, stream>>>(whhb, b_hh, wx, hbuf, hsb, cnt);

  // Phase 3: out[32768,256] = hsb[32768,512] @ regwb^T + reg_b
  gemm_bf16_kernel<<<dim3(OO / 64, (NB * TT) / 32), 256, 0, stream>>>(
      hsb, regwb, regb, out, NB * TT, OO, HH);
}
